// CognitionModel_58420145160288
// MI455X (gfx1250) — matrix operand order repaired and checked by reading, not compiled, not run on hardware
//
#include <hip/hip_runtime.h>
#include <hip/hip_bf16.h>
#include <math.h>

#define Dh 1024
#define Tt 512
#define Bb 8
#define Vv 8
#define Ll 4
#define Ii 1032
#define KP 1056   // Ii padded to multiple of 32
#define H3 3072

typedef __bf16 bf16_t;
typedef bf16_t v16bf __attribute__((ext_vector_type(16)));
typedef float  v8f   __attribute__((ext_vector_type(8)));
typedef unsigned int v4u __attribute__((ext_vector_type(4)));

union Frag16 { v16bf v; v4u q[2]; };

// ---------------------------------------------------------------------------
// CDNA5 async global->LDS copy (ASYNCcnt-tracked) + wait, via inline asm
// (portable across ROCm-7.2 / amdgpu-toolchain builtin-arity differences).
// LDS byte offset = low 32 bits of the generic shared-memory pointer.
// ---------------------------------------------------------------------------
__device__ __forceinline__ void async_copy_b128(unsigned lds_off, const void* gaddr) {
    asm volatile("global_load_async_to_lds_b128 %0, %1, off"
                 :: "v"(lds_off), "v"(gaddr)
                 : "memory");
}
__device__ __forceinline__ void wait_async0() {
    asm volatile("s_wait_asynccnt 0x0" ::: "memory");
}
__device__ __forceinline__ unsigned lds_off_of(const void* p) {
    return (unsigned)(size_t)p;
}

// ---------------------------------------------------------------------------
// WMMA fragment loaders (CDNA5 16x16x32 bf16 layouts, row-major sources)
// ---------------------------------------------------------------------------
// A (MxK tile, 16x32): lane = M (lane&15); lane<16 holds K [k0..k0+7],[k0+16..k0+23],
//                      lane>=16 holds K [k0+8..k0+15],[k0+24..k0+31]
__device__ __forceinline__ v16bf load_frag_a(const bf16_t* base, int ld, int m0, int k0, int lane) {
    int m = m0 + (lane & 15);
    int s = (lane >> 4) * 8;
    const bf16_t* p = base + (size_t)m * ld + k0;
    Frag16 f;
    f.q[0] = *(const v4u*)(p + s);
    f.q[1] = *(const v4u*)(p + 16 + s);
    return f.v;
}
// B (KxN tile, 32x16) from W stored (N,K) row-major: lane = N (lane&15);
// lane<16 holds K [k0..k0+7],[k0+16..k0+23], lane>=16 holds K [k0+8..k0+15],[k0+24..k0+31]
__device__ __forceinline__ v16bf load_frag_b(const bf16_t* base, int ld, int n0, int k0, int lane) {
    int n = n0 + (lane & 15);
    int s = (lane >> 4) * 8;
    const bf16_t* p = base + (size_t)n * ld + k0;
    Frag16 f;
    f.q[0] = *(const v4u*)(p + s);
    f.q[1] = *(const v4u*)(p + 16 + s);
    return f.v;
}

__device__ __forceinline__ v8f wmma_bf16(v16bf a, v16bf b, v8f c) {
    return __builtin_amdgcn_wmma_f32_16x16x32_bf16(false, a, false, b, (short)0, c, false, false);
}

__device__ __forceinline__ float sigmoidf_(float x) { return 1.f / (1.f + __expf(-x)); }

// ---------------------------------------------------------------------------
// 1) mask x, keep f32 copy and bf16 copy
// ---------------------------------------------------------------------------
__global__ void k_prep_x(const float* __restrict__ x, const float* __restrict__ mask,
                         float* __restrict__ xm, bf16_t* __restrict__ xbf, int n) {
    int i = blockIdx.x * blockDim.x + threadIdx.x;
    if (i >= n) return;
    float v = x[i] * mask[i / Dh];
    xm[i] = v;
    xbf[i] = (bf16_t)v;
}

// generic f32 -> bf16
__global__ void k_cvt(const float* __restrict__ s, bf16_t* __restrict__ d, int n) {
    int i = blockIdx.x * blockDim.x + threadIdx.x;
    if (i < n) d[i] = (bf16_t)s[i];
}

// op_W1 (V,D,Ii) -> bf16 padded (V,D,KP)
__global__ void k_cvt_w1(const float* __restrict__ s, bf16_t* __restrict__ d, int n) {
    int i = blockIdx.x * blockDim.x + threadIdx.x;
    if (i >= n) return;
    int row = i / KP, k = i % KP;
    d[i] = (k < Ii) ? (bf16_t)s[(size_t)row * Ii + k] : (bf16_t)0.f;
}

// ---------------------------------------------------------------------------
// 2) gx = xbf @ Wih^T + bih   (4096x1024) x (3072x1024)^T -> (4096x3072) f32
// ---------------------------------------------------------------------------
__global__ void __launch_bounds__(128) k_gemm_gx(const bf16_t* __restrict__ A,
                                                 const bf16_t* __restrict__ W,
                                                 const float* __restrict__ bias,
                                                 float* __restrict__ C) {
    int lane = threadIdx.x & 31, wave = threadIdx.x >> 5;
    int n0 = (blockIdx.x * 4 + wave) * 16;
    int m0 = blockIdx.y * 16;
    v8f acc = {};
    for (int k0 = 0; k0 < Dh; k0 += 32) {
        __builtin_prefetch(W + (size_t)(n0 + (lane & 15)) * Dh + k0 + 128, 0, 1);
        v16bf a = load_frag_a(A, Dh, m0, k0, lane);
        v16bf b = load_frag_b(W, Dh, n0, k0, lane);
        acc = wmma_bf16(a, b, acc);
    }
    int n = n0 + (lane & 15);
    float bs = bias[n];
    int mb = m0 + 8 * (lane >> 4);
#pragma unroll
    for (int r = 0; r < 8; ++r) C[(size_t)(mb + r) * H3 + n] = acc[r] + bs;
}

// ---------------------------------------------------------------------------
// 3) persistent GRU recurrence: one workgroup (32 waves), h resident in LDS.
//    Per step: async-DMA the step's gx rows into LDS (overlapped with the
//    first WMMA chunk), then per gate-chunk g: gh = h @ Whh_g^T via WMMA,
//    then gate math + h update in f32, all against LDS-resident data.
// ---------------------------------------------------------------------------
__global__ void __launch_bounds__(1024) k_gru(const float* __restrict__ gx,
                                              const bf16_t* __restrict__ Whh,
                                              const float* __restrict__ bhh,
                                              const float* __restrict__ xm,
                                              float* __restrict__ xr) {
    extern __shared__ char smem[];
    float*  h   = (float*)(smem);                 // 8x1024 f32
    bf16_t* hb  = (bf16_t*)(smem + 32 * 1024);    // 16x1024 bf16 (rows 8..15 = 0)
    float*  rg  = (float*)(smem + 64 * 1024);     // 8x1024 r gate
    float*  zg  = (float*)(smem + 96 * 1024);     // 8x1024 z gate
    float*  gh  = (float*)(smem + 128 * 1024);    // 8x1024 current chunk
    float*  gxl = (float*)(smem + 160 * 1024);    // 8x3072 f32 step-local gx
    int tid = threadIdx.x, lane = tid & 31, wave = tid >> 5;

    for (int i = tid; i < 8 * Dh; i += 1024) h[i] = 0.f;
    for (int i = tid; i < 16 * Dh; i += 1024) hb[i] = (bf16_t)0.f;
    __syncthreads();

    const unsigned gxl_off = lds_off_of(gxl);
    const int ROWCH = H3 * 4 / 16;                // 768 x 16B chunks per batch row

    for (int t = 0; t < Tt; ++t) {
        // async-stage gx rows of this step (6144 chunks / 1024 threads = 6 each)
        for (int i = tid; i < Bb * ROWCH; i += 1024) {
            int b = i / ROWCH, c = i % ROWCH;
            const void* ga = (const char*)(gx + ((size_t)b * Tt + t) * H3) + (size_t)c * 16;
            async_copy_b128(gxl_off + (unsigned)(b * H3 * 4 + c * 16), ga);
        }
        for (int g = 0; g < 3; ++g) {
            // 64 column tiles of 16; wave handles 2
            for (int tt = 0; tt < 2; ++tt) {
                int n0 = (wave * 2 + tt) * 16;
                v8f acc = {};
                const bf16_t* Wg = Whh + (size_t)g * Dh * Dh;
                for (int k0 = 0; k0 < Dh; k0 += 32) {
                    v16bf a = load_frag_a(hb, Dh, 0, k0, lane);
                    v16bf b = load_frag_b(Wg, Dh, n0, k0, lane);
                    acc = wmma_bf16(a, b, acc);
                }
                if (lane < 16) {   // rows 0..7 are the real batch rows
                    int col = n0 + lane;
#pragma unroll
                    for (int r = 0; r < 8; ++r) gh[r * Dh + col] = acc[r];
                }
            }
            if (g == 0) wait_async0();   // gx DMA covered by first WMMA chunk
            __syncthreads();
            int col = tid;
            if (g == 0) {
                for (int b = 0; b < Bb; ++b) {
                    float v = gxl[b * H3 + col] + gh[b * Dh + col] + bhh[col];
                    rg[b * Dh + col] = sigmoidf_(v);
                }
            } else if (g == 1) {
                for (int b = 0; b < Bb; ++b) {
                    float v = gxl[b * H3 + Dh + col] + gh[b * Dh + col] + bhh[Dh + col];
                    zg[b * Dh + col] = sigmoidf_(v);
                }
            } else {
                for (int b = 0; b < Bb; ++b) {
                    size_t row = (size_t)b * Tt + t;
                    float hn = gh[b * Dh + col] + bhh[2 * Dh + col];
                    float nn = tanhf(gxl[b * H3 + 2 * Dh + col] + rg[b * Dh + col] * hn);
                    float z  = zg[b * Dh + col];
                    float hnew = (1.f - z) * nn + z * h[b * Dh + col];
                    h[b * Dh + col]  = hnew;
                    hb[b * Dh + col] = (bf16_t)hnew;
                    xr[row * Dh + col] = hnew + xm[row * Dh + col];  // residual
                }
            }
            __syncthreads();
        }
    }
}

// ---------------------------------------------------------------------------
// 4) concept heads: tt (B*T,6), pol (B*T,2)
// ---------------------------------------------------------------------------
__global__ void k_concepts(const float* __restrict__ xr,
                           const float* __restrict__ ttW, const float* __restrict__ ttb,
                           const float* __restrict__ pW,  const float* __restrict__ pb,
                           float* __restrict__ tt, float* __restrict__ pol) {
    int i = blockIdx.x * blockDim.x + threadIdx.x;
    if (i >= Bb * Tt * 8) return;
    int row = i >> 3, j = i & 7;
    const float* xp = xr + (size_t)row * Dh;
    const float* w  = (j < 6) ? ttW + j * Dh : pW + (j - 6) * Dh;
    float s = (j < 6) ? ttb[j] : pb[j - 6];
    for (int k = 0; k < Dh; ++k) s += xp[k] * w[k];
    if (j < 6) tt[row * 6 + j] = s; else pol[row * 2 + (j - 6)] = s;
}

// 5) context = [mean_T(xr), mean_T(tt), mean_T(pol)]  -> (B,1032)
__global__ void k_pool_ctx(const float* __restrict__ xr, const float* __restrict__ tt,
                           const float* __restrict__ pol, float* __restrict__ ctx) {
    int i = blockIdx.x * blockDim.x + threadIdx.x;
    if (i >= Bb * Ii) return;
    int b = i / Ii, j = i % Ii;
    float s = 0.f;
    if (j < Dh)          { for (int t = 0; t < Tt; ++t) s += xr[((size_t)b * Tt + t) * Dh + j]; }
    else if (j < Dh + 6) { int c = j - Dh;      for (int t = 0; t < Tt; ++t) s += tt[(b * Tt + t) * 6 + c]; }
    else                 { int c = j - Dh - 6;  for (int t = 0; t < Tt; ++t) s += pol[(b * Tt + t) * 2 + c]; }
    ctx[i] = s / (float)Tt;
}

// 6) pg_h0 = ctx @ c2h_W^T + c2h_b
__global__ void k_pgh0(const float* __restrict__ ctx, const float* __restrict__ W,
                       const float* __restrict__ bias, float* __restrict__ h0) {
    int i = blockIdx.x * blockDim.x + threadIdx.x;
    if (i >= Bb * Dh) return;
    int b = i >> 10, d = i & (Dh - 1);
    float s = bias[d];
    const float* w = W + (size_t)d * Ii;
    const float* c = ctx + b * Ii;
    for (int k = 0; k < Ii; ++k) s += w[k] * c[k];
    h0[i] = s;
}

// 7) program-generator GRU (inputs are zero so gx == b_ih); 4 steps, single block
__global__ void __launch_bounds__(1024) k_pg_gru(const float* __restrict__ h0,
                                                 const float* __restrict__ Whh,
                                                 const float* __restrict__ bih,
                                                 const float* __restrict__ bhh,
                                                 float* __restrict__ rnn_out) {
    extern __shared__ char smem[];
    float* h  = (float*)smem;           // 8x1024
    float* h2 = h + Bb * Dh;            // 8x1024
    int col = threadIdx.x;
    for (int b = 0; b < Bb; ++b) h[b * Dh + col] = h0[b * Dh + col];
    __syncthreads();
    for (int l = 0; l < Ll; ++l) {
        for (int b = 0; b < Bb; ++b) {
            const float* hb = h + b * Dh;
            float hr = 0.f, hz = 0.f, hn = 0.f;
            const float* wr = Whh + (size_t)col * Dh;
            const float* wz = Whh + (size_t)(Dh + col) * Dh;
            const float* wn = Whh + (size_t)(2 * Dh + col) * Dh;
            for (int k = 0; k < Dh; ++k) {
                float hv = hb[k];
                hr += wr[k] * hv; hz += wz[k] * hv; hn += wn[k] * hv;
            }
            float r = sigmoidf_(bih[col] + hr + bhh[col]);
            float z = sigmoidf_(bih[Dh + col] + hz + bhh[Dh + col]);
            float n = tanhf(bih[2 * Dh + col] + r * (hn + bhh[2 * Dh + col]));
            float hnew = (1.f - z) * n + z * hb[col];
            h2[b * Dh + col] = hnew;
            rnn_out[((size_t)b * Ll + l) * Dh + col] = hnew;
        }
        __syncthreads();
        for (int b = 0; b < Bb; ++b) h[b * Dh + col] = h2[b * Dh + col];
        __syncthreads();
    }
}

// 8) program logits (B,L,V)
__global__ void k_logits(const float* __restrict__ rnn_out, const float* __restrict__ W,
                         const float* __restrict__ bias, float* __restrict__ logits) {
    int i = blockIdx.x * blockDim.x + threadIdx.x;
    if (i >= Bb * Ll * Vv) return;
    int b = i >> 5, l = (i >> 3) & 3, v = i & 7;
    float s = bias[v];
    const float* h = rnn_out + ((size_t)b * Ll + l) * Dh;
    const float* w = W + (size_t)v * Dh;
    for (int k = 0; k < Dh; ++k) s += h[k] * w[k];
    logits[(b * Ll + l) * Vv + v] = s;
}

// 9) w = mean_L softmax_V(logits)
__global__ void k_softmax_w(const float* __restrict__ logits, float* __restrict__ w) {
    int b = threadIdx.x;
    if (b >= Bb) return;
    float acc[Vv];
    for (int v = 0; v < Vv; ++v) acc[v] = 0.f;
    for (int l = 0; l < Ll; ++l) {
        const float* lg = logits + (b * Ll + l) * Vv;
        float mx = lg[0];
        for (int v = 1; v < Vv; ++v) mx = fmaxf(mx, lg[v]);
        float e[Vv], s = 0.f;
        for (int v = 0; v < Vv; ++v) { e[v] = __expf(lg[v] - mx); s += e[v]; }
        for (int v = 0; v < Vv; ++v) acc[v] += e[v] / s;
    }
    for (int v = 0; v < Vv; ++v) w[b * Vv + v] = acc[v] / (float)Ll;
}

// 10) token feats bf16 padded: [xr | tt | pol | 0]  (B*T, KP)
__global__ void k_tf(const float* __restrict__ xr, const float* __restrict__ tt,
                     const float* __restrict__ pol, bf16_t* __restrict__ o, int n) {
    int i = blockIdx.x * blockDim.x + threadIdx.x;
    if (i >= n) return;
    int row = i / KP, c = i % KP;
    float v;
    if (c < Dh)           v = xr[(size_t)row * Dh + c];
    else if (c < Dh + 6)  v = tt[row * 6 + (c - Dh)];
    else if (c < Ii)      v = pol[row * 2 + (c - Dh - 6)];
    else                  v = 0.f;
    o[i] = (bf16_t)v;
}

// ---------------------------------------------------------------------------
// 11) fused op library: per (16-token tile, v):
//     h1 = gelu(tf @ W1_v^T + b1_v) -> bf16 LDS; out += w[b,v]*(h1 @ W2_v^T + b2_v)
//     token tile staged via async global->LDS DMA.
// ---------------------------------------------------------------------------
__global__ void __launch_bounds__(256) k_oplib(const bf16_t* __restrict__ tf,
                                               const bf16_t* __restrict__ W1,
                                               const float* __restrict__ b1,
                                               const bf16_t* __restrict__ W2,
                                               const float* __restrict__ b2,
                                               const float* __restrict__ wv,
                                               float* __restrict__ out) {
    extern __shared__ char smem[];
    bf16_t* tfl = (bf16_t*)smem;                    // 16 x KP bf16 (33792 B)
    bf16_t* h1  = (bf16_t*)(smem + 16 * KP * 2);    // 16 x 1024 bf16
    int tid = threadIdx.x, lane = tid & 31, wave = tid >> 5;
    int r0 = blockIdx.x * 16;
    int v  = blockIdx.y;

    { // async-stage token tile (contiguous 16*KP bf16 block, 2112 x 16B chunks)
        const unsigned tfl_off = lds_off_of(tfl);
        const char* g = (const char*)(tf + (size_t)r0 * KP);
        for (int i = tid; i < 16 * KP * 2 / 16; i += 256)
            async_copy_b128(tfl_off + (unsigned)(i * 16), g + (size_t)i * 16);
        wait_async0();
    }
    __syncthreads();

    const bf16_t* W1v = W1 + (size_t)v * Dh * KP;
    const bf16_t* W2v = W2 + (size_t)v * Dh * Dh;

    for (int tile = wave; tile < 64; tile += 8) {
        int n0 = tile * 16;
        v8f acc = {};
        for (int k0 = 0; k0 < KP; k0 += 32) {
            v16bf a = load_frag_a(tfl, KP, 0, k0, lane);
            v16bf b = load_frag_b(W1v, KP, n0, k0, lane);
            acc = wmma_bf16(a, b, acc);
        }
        int col = n0 + (lane & 15);
        float bias = b1[v * Dh + col];
        int mb = 8 * (lane >> 4);
#pragma unroll
        for (int r = 0; r < 8; ++r) {
            float x = acc[r] + bias;
            float g = 0.5f * x * (1.f + erff(x * 0.70710678118654752f)); // exact GELU
            h1[(mb + r) * Dh + col] = (bf16_t)g;
        }
    }
    __syncthreads();

    float wb = wv[(r0 / Tt) * Vv + v];   // all 16 rows share one batch index
    for (int tile = wave; tile < 64; tile += 8) {
        int n0 = tile * 16;
        v8f acc = {};
        for (int k0 = 0; k0 < Dh; k0 += 32) {
            v16bf a = load_frag_a(h1, Dh, 0, k0, lane);
            v16bf b = load_frag_b(W2v, Dh, n0, k0, lane);
            acc = wmma_bf16(a, b, acc);
        }
        int col = n0 + (lane & 15);
        float bias = b2[v * Dh + col];
        int mb = 8 * (lane >> 4);
#pragma unroll
        for (int r = 0; r < 8; ++r)
            atomicAdd(&out[(size_t)(r0 + mb + r) * Dh + col], wb * (acc[r] + bias));
    }
}

// 12) max/mean pool over T -> pooled2 (B, 2048)
__global__ void k_pool2(const float* __restrict__ out, float* __restrict__ pooled2) {
    int i = blockIdx.x * blockDim.x + threadIdx.x;
    if (i >= Bb * Dh) return;
    int b = i >> 10, d = i & (Dh - 1);
    float mx = -INFINITY, s = 0.f;
    for (int t = 0; t < Tt; ++t) {
        float v = out[((size_t)b * Tt + t) * Dh + d];
        mx = fmaxf(mx, v); s += v;
    }
    pooled2[b * 2048 + d] = mx;
    pooled2[b * 2048 + Dh + d] = s / (float)Tt;
}

// 13) z = pooled2 @ out_W^T + out_b  -> d_out[0..8191]
__global__ void k_outproj(const float* __restrict__ pooled2, const float* __restrict__ W,
                          const float* __restrict__ bias, float* __restrict__ z) {
    int i = blockIdx.x * blockDim.x + threadIdx.x;
    if (i >= Bb * Dh) return;
    int b = i >> 10, d = i & (Dh - 1);
    float s = bias[d];
    const float* w = W + (size_t)d * 2048;
    const float* p = pooled2 + b * 2048;
    for (int k = 0; k < 2048; ++k) s += w[k] * p[k];
    z[i] = s;
}

// 14) reuse + contrastive losses over the 8x8 batch pairs
__global__ void k_loss(const float* __restrict__ pooled2, const float* __restrict__ logits,
                       float* __restrict__ dout) {
    __shared__ float pn[Bb], sq[Bb], pln[Bb];
    __shared__ float reuse[64], closs[64], pcnt[64];
    int t = threadIdx.x;
    if (t < Bb) {
        float s = 0.f;
        for (int k = 0; k < 2048; ++k) { float x = pooled2[t * 2048 + k]; s += x * x; }
        pn[t] = fmaxf(sqrtf(s), 1e-12f);
        float s2 = 0.f;
        for (int k = 0; k < Ll * Vv; ++k) { float x = logits[t * Ll * Vv + k]; s2 += x * x; }
        sq[t] = s2;
        pln[t] = fmaxf(sqrtf(s2), 1e-12f);
    }
    __syncthreads();
    int i = t >> 3, j = t & 7;
    float dp = 0.f;
    for (int k = 0; k < 2048; ++k) dp += pooled2[i * 2048 + k] * pooled2[j * 2048 + k];
    float sim = dp / (pn[i] * pn[j]);
    float dl = 0.f;
    for (int k = 0; k < Ll * Vv; ++k) dl += logits[i * Ll * Vv + k] * logits[j * Ll * Vv + k];
    float d2 = fmaxf(sq[i] + sq[j] - 2.f * dl, 0.f);
    float dist = (d2 > 0.f) ? sqrtf(d2) : 0.f;
    reuse[t] = sim * dist;
    float psim = dl / (pln[i] * pln[j]);
    float pos = (sim > 0.7f) ? 1.f : 0.f;
    float neg = (sim < 0.7f) ? 1.f : 0.f;
    closs[t] = (1.f - psim) * pos + fmaxf(psim - 0.5f, 0.f) * neg;
    pcnt[t] = pos + neg;
    __syncthreads();
    if (t == 0) {
        float rs = 0.f, cs = 0.f, cn = 0.f;
        for (int k = 0; k < 64; ++k) { rs += reuse[k]; cs += closs[k]; cn += pcnt[k]; }
        dout[Bb * Dh]     = rs / 64.f;
        dout[Bb * Dh + 1] = cs / (cn + 1e-6f);
    }
}

// ---------------------------------------------------------------------------
extern "C" void kernel_launch(void* const* d_in, const int* in_sizes, int n_in,
                              void* d_out, int out_size, void* d_ws, size_t ws_size,
                              hipStream_t stream) {
    const float* x        = (const float*)d_in[0];
    const float* amask    = (const float*)d_in[1];
    const float* enc_Wih  = (const float*)d_in[2];
    const float* enc_Whh  = (const float*)d_in[3];
    const float* enc_bih  = (const float*)d_in[4];
    const float* enc_bhh  = (const float*)d_in[5];
    const float* tt_W     = (const float*)d_in[6];
    const float* tt_b     = (const float*)d_in[7];
    const float* pol_W    = (const float*)d_in[8];
    const float* pol_b    = (const float*)d_in[9];
    const float* c2h_W    = (const float*)d_in[10];
    const float* c2h_b    = (const float*)d_in[11];
    // d_in[12] = pg_Wih (unused: GRU inputs are zero)
    const float* pg_Whh   = (const float*)d_in[13];
    const float* pg_bih   = (const float*)d_in[14];
    const float* pg_bhh   = (const float*)d_in[15];
    const float* proj_W   = (const float*)d_in[16];
    const float* proj_b   = (const float*)d_in[17];
    const float* op_W1    = (const float*)d_in[18];
    const float* op_b1    = (const float*)d_in[19];
    const float* op_W2    = (const float*)d_in[20];
    const float* op_b2    = (const float*)d_in[21];
    const float* out_W    = (const float*)d_in[22];
    const float* out_b    = (const float*)d_in[23];
    float* dout = (float*)d_out;

    // workspace carve-out
    char* ws = (char*)d_ws;
    size_t off = 0;
    auto alloc = [&](size_t bytes) -> void* {
        void* p = ws + off;
        off += (bytes + 255) & ~(size_t)255;
        return p;
    };
    const int NTOK = Bb * Tt;                       // 4096
    float*  xm      = (float*) alloc((size_t)NTOK * Dh * 4);
    bf16_t* xbf     = (bf16_t*)alloc((size_t)NTOK * Dh * 2);
    bf16_t* wih_bf  = (bf16_t*)alloc((size_t)H3 * Dh * 2);
    bf16_t* whh_bf  = (bf16_t*)alloc((size_t)H3 * Dh * 2);
    float*  gx      = (float*) alloc((size_t)NTOK * H3 * 4);
    float*  xr      = (float*) alloc((size_t)NTOK * Dh * 4);
    float*  ttv     = (float*) alloc((size_t)NTOK * 6 * 4);
    float*  polv    = (float*) alloc((size_t)NTOK * 2 * 4);
    float*  ctx     = (float*) alloc((size_t)Bb * Ii * 4);
    float*  pgh0    = (float*) alloc((size_t)Bb * Dh * 4);
    float*  rnn_out = (float*) alloc((size_t)Bb * Ll * Dh * 4);
    float*  logits  = (float*) alloc((size_t)Bb * Ll * Vv * 4);
    float*  wv      = (float*) alloc((size_t)Bb * Vv * 4);
    bf16_t* tf_bf   = (bf16_t*)alloc((size_t)NTOK * KP * 2);
    bf16_t* w1_bf   = (bf16_t*)alloc((size_t)Vv * Dh * KP * 2);
    bf16_t* w2_bf   = (bf16_t*)alloc((size_t)Vv * Dh * Dh * 2);
    float*  opout   = (float*) alloc((size_t)NTOK * Dh * 4);
    float*  pooled2 = (float*) alloc((size_t)Bb * 2048 * 4);

    // opt-in to large dynamic LDS (320 KB per WGP on CDNA5)
    (void)hipFuncSetAttribute((const void*)k_gru,    hipFuncAttributeMaxDynamicSharedMemorySize, 256 * 1024);
    (void)hipFuncSetAttribute((const void*)k_pg_gru, hipFuncAttributeMaxDynamicSharedMemorySize, 64 * 1024);
    (void)hipFuncSetAttribute((const void*)k_oplib,  hipFuncAttributeMaxDynamicSharedMemorySize, 68 * 1024);

    // 1) prep
    {
        int n = NTOK * Dh;
        k_prep_x<<<(n + 255) / 256, 256, 0, stream>>>(x, amask, xm, xbf, n);
    }
    // weight conversions
    {
        int n = H3 * Dh;
        k_cvt<<<(n + 255) / 256, 256, 0, stream>>>(enc_Wih, wih_bf, n);
        k_cvt<<<(n + 255) / 256, 256, 0, stream>>>(enc_Whh, whh_bf, n);
        int n2 = Vv * Dh * Dh;
        k_cvt<<<(n2 + 255) / 256, 256, 0, stream>>>(op_W2, w2_bf, n2);
        int n1 = Vv * Dh * KP;
        k_cvt_w1<<<(n1 + 255) / 256, 256, 0, stream>>>(op_W1, w1_bf, n1);
    }
    // 2) gx GEMM: tiles 192(N) x 256(M), 4 waves/block
    k_gemm_gx<<<dim3(48, 256), 128, 0, stream>>>(xbf, wih_bf, enc_bih, gx);
    // 3) persistent GRU (single WGP, 256 KB LDS incl. async-staged gx)
    k_gru<<<1, 1024, 256 * 1024, stream>>>(gx, whh_bf, enc_bhh, xm, xr);
    // 4) concept heads
    {
        int n = NTOK * 8;
        k_concepts<<<(n + 255) / 256, 256, 0, stream>>>(xr, tt_W, tt_b, pol_W, pol_b, ttv, polv);
    }
    // 5) context
    {
        int n = Bb * Ii;
        k_pool_ctx<<<(n + 255) / 256, 256, 0, stream>>>(xr, ttv, polv, ctx);
    }
    // 6) pg_h0
    k_pgh0<<<(Bb * Dh + 255) / 256, 256, 0, stream>>>(ctx, c2h_W, c2h_b, pgh0);
    // 7) program-generator GRU
    k_pg_gru<<<1, 1024, 64 * 1024, stream>>>(pgh0, pg_Whh, pg_bih, pg_bhh, rnn_out);
    // 8) logits
    k_logits<<<1, 256, 0, stream>>>(rnn_out, proj_W, proj_b, logits);
    // 9) softmax weights
    k_softmax_w<<<1, 32, 0, stream>>>(logits, wv);
    // 10) token feats (bf16, padded)
    {
        int n = NTOK * KP;
        k_tf<<<(n + 255) / 256, 256, 0, stream>>>(xr, ttv, polv, tf_bf, n);
    }
    // 11) fused op library (zero accumulator first)
    (void)hipMemsetAsync(opout, 0, (size_t)NTOK * Dh * 4, stream);
    k_oplib<<<dim3(NTOK / 16, Vv), 256, 16 * KP * 2 + 16 * Dh * 2, stream>>>(
        tf_bf, w1_bf, op_b1, w2_bf, op_b2, wv, opout);
    // 12) pooling
    k_pool2<<<(Bb * Dh + 255) / 256, 256, 0, stream>>>(opout, pooled2);
    // 13) output projection -> d_out[0..8191]
    k_outproj<<<(Bb * Dh + 255) / 256, 256, 0, stream>>>(pooled2, out_W, out_b, dout);
    // 14) losses -> d_out[8192], d_out[8193]
    k_loss<<<1, 64, 0, stream>>>(pooled2, logits, dout);
}
